// Decoder_23330262352067
// MI455X (gfx1250) — matrix operand order repaired and checked by reading, not compiled, not run on hardware
//
#include <hip/hip_runtime.h>
#include <hip/hip_bf16.h>
#include <cstdint>

// ---------------------------------------------------------------------------
// Decoder (N=1, T=2048, DIM=1024, 16 heads x 64, MLP=4096, DEPTH=4, window 512)
// bf16 WMMA everywhere for matmuls, f32 accumulation.
// ---------------------------------------------------------------------------

#define T_SEQ   2048
#define DIM     1024
#define NH      16
#define HD      64
#define MLPD    4096
#define DEPTH   4
#define WINDOW  512
#define EPSF    1e-6f

typedef __bf16  v16bf __attribute__((ext_vector_type(16)));
typedef __bf16  v8bf  __attribute__((ext_vector_type(8)));
typedef __bf16  v4bf  __attribute__((ext_vector_type(4)));
typedef float   v8f   __attribute__((ext_vector_type(8)));
typedef float   v4f   __attribute__((ext_vector_type(4)));
typedef int     v4i   __attribute__((ext_vector_type(4)));

// --------- CDNA5 async global->LDS path (guarded; fallback = reg pipeline) --
#if defined(__has_builtin)
# if __has_builtin(__builtin_amdgcn_global_load_async_to_lds_b128)
#  define HAVE_ASYNC_LDS 1
# endif
#endif
#ifndef HAVE_ASYNC_LDS
# define HAVE_ASYNC_LDS 0
#endif

#if HAVE_ASYNC_LDS
__device__ __forceinline__ void stage16_async(const __bf16* g, __bf16* l) {
    // Builtin signature: (v4i AS1*, v4i AS3*, imm offset, imm cpol)
    auto gp = (__attribute__((address_space(1))) v4i*)(unsigned long long)(uintptr_t)g;
    auto lp = (__attribute__((address_space(3))) v4i*)(unsigned int)(uintptr_t)l;
    __builtin_amdgcn_global_load_async_to_lds_b128(gp, lp, 0, 0);
}
template <int N> __device__ __forceinline__ void wait_asynccnt() {
# if __has_builtin(__builtin_amdgcn_s_wait_asynccnt)
    __builtin_amdgcn_s_wait_asynccnt((unsigned short)N);
# else
    if (N == 0) asm volatile("s_wait_asynccnt 0" ::: "memory");
    else        asm volatile("s_wait_asynccnt 6" ::: "memory");
# endif
}
#endif

__device__ __forceinline__ v16bf pack16(v8bf lo, v8bf hi) {
    v16bf r;
#pragma unroll
    for (int i = 0; i < 8; ++i) { r[i] = lo[i]; r[i + 8] = hi[i]; }
    return r;
}

__device__ __forceinline__ v8f wmma_bf16(v16bf a, v16bf b, v8f c) {
    return __builtin_amdgcn_wmma_f32_16x16x32_bf16(
        false, a, false, b, (short)0, c, false, false);
}

// ---------------------------------------------------------------------------
// Elementwise f32 -> bf16 cast (weights)
// ---------------------------------------------------------------------------
__global__ void cast_f32_bf16(const float* __restrict__ in, __bf16* __restrict__ out, int n) {
    int i = blockIdx.x * 256 + threadIdx.x;
    int stride = gridDim.x * 256;
    for (; i < n; i += stride) out[i] = (__bf16)in[i];
}

// ---------------------------------------------------------------------------
// Row RMSNorm (over DIM) + bf16 cast.  One block (256 thr) per token row.
// ---------------------------------------------------------------------------
__global__ __launch_bounds__(256) void rmsnorm_cast(const float* __restrict__ x,
                                                    __bf16* __restrict__ y) {
    const int row = blockIdx.x;
    const int tid = threadIdx.x;
    const float* xr = x + (size_t)row * DIM;
    float v[4];
    float s = 0.f;
#pragma unroll
    for (int i = 0; i < 4; ++i) { v[i] = xr[tid + 256 * i]; s += v[i] * v[i]; }
    __shared__ float red[256];
    red[tid] = s;
    __syncthreads();
    for (int off = 128; off > 0; off >>= 1) {
        if (tid < off) red[tid] += red[tid + off];
        __syncthreads();
    }
    const float scale = rsqrtf(red[0] / (float)DIM + EPSF);
#pragma unroll
    for (int i = 0; i < 4; ++i)
        y[(size_t)row * DIM + tid + 256 * i] = (__bf16)(v[i] * scale);
}

// ---------------------------------------------------------------------------
// Tiled bf16 WMMA GEMM:  C[M][N] = A[M][K] * W[N][K]^T
// Block tile 256x128, K-step 32, 8 waves -> each wave 64x64 (16 WMMA tiles).
// Async-DMA double-buffered LDS pipeline; K and epilogue MODE compile-time.
// MODE 0: outF = C ; 1: outF = residual + C ; 2: outB = bf16(relu(C)^2)
// ---------------------------------------------------------------------------
#define BM 256
#define BN 128
#define BK 32

__device__ __forceinline__ void gemm_compute(const __bf16* As, const __bf16* Bs,
                                             v8f acc[4][4],
                                             int wr, int wc, int ln, int lh) {
    v16bf af[4], bfm[4];
#pragma unroll
    for (int tm = 0; tm < 4; ++tm) {
        int rr = wr * 64 + tm * 16 + ln;
        af[tm] = *(const v16bf*)&As[rr * BK + 16 * lh];
    }
#pragma unroll
    for (int tn = 0; tn < 4; ++tn) {
        int cc = wc * 64 + tn * 16 + ln;
        bfm[tn] = *(const v16bf*)&Bs[cc * BK + 16 * lh];
    }
#pragma unroll
    for (int tm = 0; tm < 4; ++tm)
#pragma unroll
        for (int tn = 0; tn < 4; ++tn)
            acc[tm][tn] = wmma_bf16(af[tm], bfm[tn], acc[tm][tn]);
}

template <int K, int MODE>
__global__ __launch_bounds__(256) void gemm_bf16_wmma(
    const __bf16* __restrict__ A, const __bf16* __restrict__ W,
    int N, float* outF, __bf16* outB, const float* residual)
{
    __shared__ alignas(32) __bf16 As[2][BM * BK];   // 2 x 16 KB
    __shared__ alignas(32) __bf16 Bs[2][BN * BK];   // 2 x  8 KB

    const int tid  = threadIdx.x;
    const int wave = tid >> 5;
    const int lane = tid & 31;
    const int ln   = lane & 15;
    const int lh   = lane >> 4;
    const int wr   = wave & 3;   // 4 wave-rows -> 64 rows each
    const int wc   = wave >> 2;  // 2 wave-cols -> 64 cols each
    const int row0 = blockIdx.y * BM;
    const int col0 = blockIdx.x * BN;

    // Staging: 16B chunks; thread owns rows r0 + {0,64,128,192} of A and
    // rows r0 + {0,64} of W, at intra-row offset o0.
    const int r0   = tid >> 2;
    const int o0   = (tid & 3) * 8;
    const int lOff = r0 * BK + o0;
    const __bf16* gA = A + (size_t)(row0 + r0) * K + o0;
    const __bf16* gW = W + (size_t)(col0 + r0) * K + o0;
    const size_t  rstep = (size_t)64 * K;

    constexpr int NKT = K / BK;
    v8f acc[4][4] = {};
    int buf = 0;

#if HAVE_ASYNC_LDS
    // Async DMA pipeline: ASYNCcnt-tracked global->LDS, overlapped with WMMA.
#pragma unroll
    for (int i = 0; i < 4; ++i) stage16_async(gA + i * rstep, &As[0][lOff + i * 64 * BK]);
#pragma unroll
    for (int i = 0; i < 2; ++i) stage16_async(gW + i * rstep, &Bs[0][lOff + i * 64 * BK]);
#pragma unroll 1
    for (int kt = 0; kt < NKT; ++kt) {
        if (kt + 1 < NKT) {
            const __bf16* a = gA + (size_t)(kt + 1) * BK;
            const __bf16* w = gW + (size_t)(kt + 1) * BK;
#pragma unroll
            for (int i = 0; i < 4; ++i) stage16_async(a + i * rstep, &As[buf ^ 1][lOff + i * 64 * BK]);
#pragma unroll
            for (int i = 0; i < 2; ++i) stage16_async(w + i * rstep, &Bs[buf ^ 1][lOff + i * 64 * BK]);
            wait_asynccnt<6>();   // current buffer's 6 transfers complete
        } else {
            wait_asynccnt<0>();
        }
        __syncthreads();
        gemm_compute(As[buf], Bs[buf], acc, wr, wc, ln, lh);
        __syncthreads();
        buf ^= 1;
    }
#else
    // Register-staged pipeline: prefetch next K-tile into VGPRs during WMMA.
    v8bf ra[4], rw[2];
#pragma unroll
    for (int i = 0; i < 4; ++i) ra[i] = *(const v8bf*)(gA + i * rstep);
#pragma unroll
    for (int i = 0; i < 2; ++i) rw[i] = *(const v8bf*)(gW + i * rstep);
#pragma unroll 1
    for (int kt = 0; kt < NKT; ++kt) {
#pragma unroll
        for (int i = 0; i < 4; ++i) *(v8bf*)&As[buf][lOff + i * 64 * BK] = ra[i];
#pragma unroll
        for (int i = 0; i < 2; ++i) *(v8bf*)&Bs[buf][lOff + i * 64 * BK] = rw[i];
        __syncthreads();
        if (kt + 1 < NKT) {
            const __bf16* a = gA + (size_t)(kt + 1) * BK;
            const __bf16* w = gW + (size_t)(kt + 1) * BK;
#pragma unroll
            for (int i = 0; i < 4; ++i) ra[i] = *(const v8bf*)(a + i * rstep);
#pragma unroll
            for (int i = 0; i < 2; ++i) rw[i] = *(const v8bf*)(w + i * rstep);
        }
        gemm_compute(As[buf], Bs[buf], acc, wr, wc, ln, lh);
        __syncthreads();
        buf ^= 1;
    }
#endif

    // Epilogue. C tile layout: row = (lane>=16)*8 + r, col = lane&15.
#pragma unroll
    for (int tm = 0; tm < 4; ++tm)
#pragma unroll
        for (int tn = 0; tn < 4; ++tn) {
            int rbase = row0 + wr * 64 + tm * 16 + lh * 8;
            int col   = col0 + wc * 64 + tn * 16 + ln;
#pragma unroll
            for (int r = 0; r < 8; ++r) {
                size_t idx = (size_t)(rbase + r) * N + col;
                float v = acc[tm][tn][r];
                if constexpr (MODE == 0) {
                    outF[idx] = v;
                } else if constexpr (MODE == 1) {
                    outF[idx] = residual[idx] + v;
                } else {
                    float t = v > 0.f ? v * v : 0.f;
                    outB[idx] = (__bf16)t;
                }
            }
        }
}

// ---------------------------------------------------------------------------
// QKV post-processing: per-head RMSNorm(q,k), RoPE(q,k), relayout to
// Q,K : [NH][T][HD] bf16 ; V transposed: Vt[NH][HD][T] bf16.
// ---------------------------------------------------------------------------
__global__ __launch_bounds__(64) void qkv_post(const float* __restrict__ hb,
                                               const float* __restrict__ cosb,
                                               const float* __restrict__ sinb,
                                               __bf16* __restrict__ Qo,
                                               __bf16* __restrict__ Ko,
                                               __bf16* __restrict__ Vt)
{
    const int t = blockIdx.x;
    const int h = blockIdx.y;
    const int d = threadIdx.x;
    const float* base = hb + (size_t)t * (3 * DIM);
    float q = base[h * HD + d];
    float k = base[DIM + h * HD + d];
    float v = base[2 * DIM + h * HD + d];

    __shared__ float rq[64], rk[64], qs[64], ks[64];
    rq[d] = q * q; rk[d] = k * k;
    __syncthreads();
    for (int off = 32; off > 0; off >>= 1) {
        if (d < off) { rq[d] += rq[d + off]; rk[d] += rk[d + off]; }
        __syncthreads();
    }
    const float sq = rsqrtf(rq[0] / (float)HD + EPSF);
    const float sk = rsqrtf(rk[0] / (float)HD + EPSF);
    qs[d] = q * sq; ks[d] = k * sk;
    __syncthreads();

    float qo, ko;
    if (d < 32) {
        float c = cosb[t * 32 + d], s = sinb[t * 32 + d];
        qo = qs[d] * c - qs[d + 32] * s;
        ko = ks[d] * c - ks[d + 32] * s;
    } else {
        int j = d - 32;
        float c = cosb[t * 32 + j], s = sinb[t * 32 + j];
        qo = qs[j] * s + qs[d] * c;
        ko = ks[j] * s + ks[d] * c;
    }
    size_t qi = ((size_t)h * T_SEQ + t) * HD + d;
    Qo[qi] = (__bf16)qo;
    Ko[qi] = (__bf16)ko;
    Vt[((size_t)h * HD + d) * T_SEQ + t] = (__bf16)v;
}

// ---------------------------------------------------------------------------
// Sliding-window attention, one block (128 thr, 4 waves) per (qblock, head).
// ---------------------------------------------------------------------------
#define MAXKT 34   // 33 key tiles max + 1 pad tile

__global__ __launch_bounds__(128) void attn_window(const __bf16* __restrict__ Qb,
                                                   const __bf16* __restrict__ Kb,
                                                   const __bf16* __restrict__ Vt,
                                                   __bf16* __restrict__ attnb)
{
    __shared__ alignas(16) float  sc[16][MAXKT * 16];
    __shared__ alignas(16) __bf16 pr[16][MAXKT * 16];
    __shared__ float  red2[16][8];
    __shared__ float  rowstat[16];

    const int qb = blockIdx.x;
    const int h  = blockIdx.y;
    const int t0 = qb * 16;
    const int kbmin = (qb > 32) ? (qb - 32) : 0;
    const int nkb   = qb - kbmin + 1;          // <= 33 key tiles

    const int tid  = threadIdx.x;
    const int wave = tid >> 5;
    const int lane = tid & 31;
    const int ln   = lane & 15;
    const int lh   = lane >> 4;

    // --- Stage 1: scores S = QK^T ----------------------------------------
    v16bf qa[2];
#pragma unroll
    for (int s = 0; s < 2; ++s) {
        size_t base = ((size_t)h * T_SEQ + (t0 + ln)) * HD + s * 32;
        qa[s] = *(const v16bf*)(Qb + base + 16 * lh);
    }
    for (int kbi = wave; kbi < nkb; kbi += 4) {
        const int kt0 = (kbmin + kbi) * 16;
        v8f acc = {};
#pragma unroll
        for (int s = 0; s < 2; ++s) {
            v16bf kf = *(const v16bf*)(Kb + ((size_t)h * T_SEQ + (kt0 + ln)) * HD + s * 32 + 16 * lh);
            acc = wmma_bf16(qa[s], kf, acc);
        }
        const int tk = kt0 + ln;
#pragma unroll
        for (int r = 0; r < 8; ++r) {
            int rowi = lh * 8 + r;
            int tq = t0 + rowi;
            float v = acc[r] * 0.125f;                      // 1/sqrt(64)
            if (!(tk <= tq && tq < tk + WINDOW)) v = -1e30f;
            sc[rowi][kbi * 16 + ln] = v;
        }
    }
    __syncthreads();

    // --- Stage 2: softmax (vectorized: 8 lanes x float4 chunks per row) ---
    const int L   = nkb * 16;
    const int row = tid >> 3;
    const int sub = tid & 7;
    float mx = -1e30f;
    for (int j = sub * 4; j < L; j += 32) {
        v4f v = *(const v4f*)&sc[row][j];
        mx = fmaxf(mx, fmaxf(fmaxf(v[0], v[1]), fmaxf(v[2], v[3])));
    }
    red2[row][sub] = mx;
    __syncthreads();
    if (sub == 0) {
        float m = red2[row][0];
#pragma unroll
        for (int i = 1; i < 8; ++i) m = fmaxf(m, red2[row][i]);
        rowstat[row] = m;
    }
    __syncthreads();
    const float m = rowstat[row];
    float s = 0.f;
    for (int j = sub * 4; j < L; j += 32) {
        v4f v = *(const v4f*)&sc[row][j];
        v4f e;
#pragma unroll
        for (int i = 0; i < 4; ++i) { e[i] = __expf(v[i] - m); s += e[i]; }
        *(v4f*)&sc[row][j] = e;
    }
    red2[row][sub] = s;
    __syncthreads();
    if (sub == 0) {
        float t = 0.f;
#pragma unroll
        for (int i = 0; i < 8; ++i) t += red2[row][i];
        rowstat[row] = 1.f / t;
    }
    __syncthreads();
    const float inv = rowstat[row];
    for (int j = sub * 4; j < L; j += 32) {
        v4f e = *(const v4f*)&sc[row][j];
        v4bf p;
#pragma unroll
        for (int i = 0; i < 4; ++i) p[i] = (__bf16)(e[i] * inv);
        *(v4bf*)&pr[row][j] = p;
    }
    if (nkb & 1) { // zero-pad one key tile so the PV K-loop can step by 32
        for (int j = tid; j < 256; j += 128) pr[j >> 4][nkb * 16 + (j & 15)] = (__bf16)0.f;
    }
    __syncthreads();

    // --- Stage 3: O = P V  (wave w -> head-dim cols 16w..16w+15) ----------
    v8f o = {};
    const int nst = (nkb + 1) >> 1;
    for (int st = 0; st < nst; ++st) {
        v16bf pa = pack16(*(const v8bf*)&pr[ln][st * 32 + 8 * lh],
                          *(const v8bf*)&pr[ln][st * 32 + 16 + 8 * lh]);
        const __bf16* vp = Vt + ((size_t)h * HD + wave * 16 + ln) * T_SEQ + kbmin * 16 + st * 32;
        v16bf vb = pack16(*(const v8bf*)(vp + 8 * lh), *(const v8bf*)(vp + 16 + 8 * lh));
        o = wmma_bf16(pa, vb, o);
    }
    const int col = h * HD + wave * 16 + ln;
#pragma unroll
    for (int r = 0; r < 8; ++r)
        attnb[(size_t)(t0 + lh * 8 + r) * DIM + col] = (__bf16)o[r];
}

// ---------------------------------------------------------------------------
// Host-side launch
// ---------------------------------------------------------------------------
extern "C" void kernel_launch(void* const* d_in, const int* in_sizes, int n_in,
                              void* d_out, int out_size, void* d_ws, size_t ws_size,
                              hipStream_t stream) {
    (void)in_sizes; (void)n_in; (void)out_size; (void)ws_size;
    const float* x     = (const float*)d_in[0];
    const float* qkv_w = (const float*)d_in[1];
    const float* o_w   = (const float*)d_in[2];
    const float* up_w  = (const float*)d_in[3];
    const float* dn_w  = (const float*)d_in[4];
    const float* cosb  = (const float*)d_in[5];
    const float* sinb  = (const float*)d_in[6];
    float* xo = (float*)d_out;

    char* ws = (char*)d_ws;
    size_t off = 0;
    auto alloc = [&](size_t bytes) -> char* {
        char* p = ws + off;
        off += (bytes + 255) & ~(size_t)255;
        return p;
    };
    const int nWq = DEPTH * 3 * DIM * DIM;
    const int nWo = DEPTH * DIM * DIM;
    const int nWu = DEPTH * MLPD * DIM;
    __bf16* wq = (__bf16*)alloc((size_t)nWq * 2);
    __bf16* wo = (__bf16*)alloc((size_t)nWo * 2);
    __bf16* wu = (__bf16*)alloc((size_t)nWu * 2);
    __bf16* wd = (__bf16*)alloc((size_t)nWu * 2);
    __bf16* xn    = (__bf16*)alloc((size_t)T_SEQ * DIM * 2);
    float*  hbuf  = (float*)alloc((size_t)T_SEQ * 3 * DIM * 4);
    __bf16* Qb    = (__bf16*)alloc((size_t)T_SEQ * DIM * 2);
    __bf16* Kb    = (__bf16*)alloc((size_t)T_SEQ * DIM * 2);
    __bf16* Vt    = (__bf16*)alloc((size_t)T_SEQ * DIM * 2);
    __bf16* attnb = (__bf16*)alloc((size_t)T_SEQ * DIM * 2);
    __bf16* hm    = (__bf16*)alloc((size_t)T_SEQ * MLPD * 2);

    cast_f32_bf16<<<4096, 256, 0, stream>>>(qkv_w, wq, nWq);
    cast_f32_bf16<<<4096, 256, 0, stream>>>(o_w,   wo, nWo);
    cast_f32_bf16<<<4096, 256, 0, stream>>>(up_w,  wu, nWu);
    cast_f32_bf16<<<4096, 256, 0, stream>>>(dn_w,  wd, nWu);

    (void)hipMemcpyAsync(xo, x, (size_t)T_SEQ * DIM * sizeof(float),
                         hipMemcpyDeviceToDevice, stream);

    for (int l = 0; l < DEPTH; ++l) {
        // ---- attention block ----
        rmsnorm_cast<<<T_SEQ, 256, 0, stream>>>(xo, xn);
        gemm_bf16_wmma<DIM, 0><<<dim3(3 * DIM / BN, T_SEQ / BM), 256, 0, stream>>>(
            xn, wq + (size_t)l * 3 * DIM * DIM, 3 * DIM, hbuf, nullptr, nullptr);
        qkv_post<<<dim3(T_SEQ, NH), 64, 0, stream>>>(hbuf, cosb, sinb, Qb, Kb, Vt);
        attn_window<<<dim3(T_SEQ / 16, NH), 128, 0, stream>>>(Qb, Kb, Vt, attnb);
        gemm_bf16_wmma<DIM, 1><<<dim3(DIM / BN, T_SEQ / BM), 256, 0, stream>>>(
            attnb, wo + (size_t)l * DIM * DIM, DIM, xo, nullptr, xo);

        // ---- MLP block ----
        rmsnorm_cast<<<T_SEQ, 256, 0, stream>>>(xo, xn);
        gemm_bf16_wmma<DIM, 2><<<dim3(MLPD / BN, T_SEQ / BM), 256, 0, stream>>>(
            xn, wu + (size_t)l * MLPD * DIM, MLPD, nullptr, hm, nullptr);
        gemm_bf16_wmma<MLPD, 1><<<dim3(DIM / BN, T_SEQ / BM), 256, 0, stream>>>(
            hm, wd + (size_t)l * DIM * MLPD, DIM, xo, nullptr, xo);
    }
}
